// ViTSelfAttention_50835232916254
// MI455X (gfx1250) — compile-verified, matrix operand order checked by reading, not run on hardware
//
#include <hip/hip_runtime.h>
#include <hip/hip_bf16.h>

// Problem dims (fixed by the reference)
#define BB 16
#define SS 1024
#define HH 768
#define NHD 12
#define HD 64

typedef __attribute__((ext_vector_type(16))) __bf16 v16bf;
typedef __attribute__((ext_vector_type(4)))  __bf16 v4bf;
typedef __attribute__((ext_vector_type(8)))  float  v8f;
typedef __attribute__((ext_vector_type(4)))  float  f32x4;

__device__ __forceinline__ v8f wmma_bf16(v16bf a, v16bf b, v8f c) {
  // D = A(16x32 bf16) x B(32x16 bf16) + C(16x16 f32)
  return __builtin_amdgcn_wmma_f32_16x16x32_bf16(
      /*neg_a=*/false, a, /*neg_b=*/false, b,
      /*c_mod=*/(short)0, c, /*reuse_a=*/false, /*reuse_b=*/false);
}

// Async bulk copy: one 16-byte segment per lane, global -> LDS (ASYNCcnt).
__device__ __forceinline__ void async_copy_b128(unsigned lds_addr,
                                                const void* gaddr) {
  asm volatile("global_load_async_to_lds_b128 %0, %1, off"
               :: "v"(lds_addr), "v"(gaddr) : "memory");
}
__device__ __forceinline__ void wait_async0() {
  asm volatile("s_wait_asynccnt 0" ::: "memory");
}

// ---------------------------------------------------------------------------
// Kernel 0: one-shot fp32 -> bf16 conversion (weights). n4 = elements/4.
// ---------------------------------------------------------------------------
__global__ __launch_bounds__(256)
void cvt_f32_to_bf16_kernel(const float* __restrict__ in,
                            __bf16* __restrict__ out, int n4) {
  int idx = blockIdx.x * 256 + threadIdx.x;
  if (idx < n4) {
    f32x4 t = ((const f32x4*)in)[idx];
    v4bf b;
    b[0] = (__bf16)t[0]; b[1] = (__bf16)t[1];
    b[2] = (__bf16)t[2]; b[3] = (__bf16)t[3];
    ((v4bf*)out)[idx] = b;
  }
}

// ---------------------------------------------------------------------------
// One projection y = A_tile @ W^T + b. Force-inlined with *direct* kernel-arg
// pointers so address-space inference yields global_load (not FLAT), and each
// of the three phases is register-allocated independently.
// TRANSPOSED=false: out [B,NH,S,HD].  TRANSPOSED=true: out [B,NH,HD,S].
// ---------------------------------------------------------------------------
template <bool TRANSPOSED>
__device__ __forceinline__ void proj_one(const __bf16* __restrict__ W,
                                         const float* __restrict__ bias,
                                         __bf16* __restrict__ out,
                                         const __bf16 (&sA)[32][HH],
                                         int nbase, int lr, int lh,
                                         int bidx, int srow0) {
  v8f acc[2][6] = {};

  for (int k0 = 0; k0 < HH; k0 += 32) {
    v16bf a0 = *(const v16bf*)&sA[lr][k0 + lh * 16];
    v16bf a1 = *(const v16bf*)&sA[16 + lr][k0 + lh * 16];
#pragma unroll
    for (int t = 0; t < 6; ++t) {
      v16bf b = *(const v16bf*)(W + (size_t)(nbase + t * 16 + lr) * HH +
                                k0 + lh * 16);
      acc[0][t] = wmma_bf16(a0, b, acc[0][t]);
      acc[1][t] = wmma_bf16(a1, b, acc[1][t]);
    }
  }

  // Epilogue: bias add + scatter
#pragma unroll
  for (int t = 0; t < 6; ++t) {
    int n = nbase + t * 16 + lr;
    float bv = bias[n];
    int h = n / HD, d = n % HD;
    int hoff = bidx * NHD + h;
#pragma unroll
    for (int mi = 0; mi < 2; ++mi) {
#pragma unroll
      for (int r = 0; r < 8; ++r) {
        int srow = srow0 + mi * 16 + r + 8 * lh;
        __bf16 val = (__bf16)(acc[mi][t][r] + bv);
        if (TRANSPOSED)
          out[(size_t)(hoff * HD + d) * SS + srow] = val;
        else
          out[(size_t)(hoff * SS + srow) * HD + d] = val;
      }
    }
  }
}

// ---------------------------------------------------------------------------
// Kernel 1: fused QKV projection (bf16 weights pre-converted). One 32-row M
// tile per block, staged once in LDS and reused for Q,K,V. 8 waves; wave owns
// 2 M-subtiles x 6 N-tiles = 12 independent WMMA chains per k-step.
// ---------------------------------------------------------------------------
__global__ __launch_bounds__(256)
void qkv_proj_kernel(const float* __restrict__ hs,
                     const __bf16* __restrict__ wq, const __bf16* __restrict__ wk,
                     const __bf16* __restrict__ wv,
                     const float* __restrict__ bq, const float* __restrict__ bk,
                     const float* __restrict__ bv,
                     __bf16* __restrict__ qo, __bf16* __restrict__ ko,
                     __bf16* __restrict__ vo) {
  __shared__ __bf16 sA[32][HH];   // 48 KB activation tile (bf16)

  const int tid   = threadIdx.x;
  const int mbase = blockIdx.x * 32;

  // Stage 32x768 fp32 -> bf16 into LDS once (6144 float4 / 256 thr = 24 each)
  const float* arow = hs + (size_t)mbase * HH;
#pragma unroll
  for (int i = 0; i < 24; ++i) {
    int flat = tid + i * 256;        // float4 index within tile
    int r = flat / 192, c4 = flat % 192;
    f32x4 t = *(const f32x4*)(arow + r * HH + c4 * 4);
    v4bf b4;
    b4[0] = (__bf16)t[0]; b4[1] = (__bf16)t[1];
    b4[2] = (__bf16)t[2]; b4[3] = (__bf16)t[3];
    *(v4bf*)&sA[r][c4 * 4] = b4;
  }
  __syncthreads();

  const int lane  = tid & 31;
  const int wave  = tid >> 5;
  const int nbase = wave * 96;
  const int lr    = lane & 15;
  const int lh    = lane >> 4;
  const int bidx  = mbase / SS;
  const int srow0 = mbase % SS;

  proj_one<false>(wq, bq, qo, sA, nbase, lr, lh, bidx, srow0);
  proj_one<false>(wk, bk, ko, sA, nbase, lr, lh, bidx, srow0);
  proj_one<true >(wv, bv, vo, sA, nbase, lr, lh, bidx, srow0);  // V transposed
}

// ---------------------------------------------------------------------------
// Kernel 2: fused flash attention. Q,K in [B,NH,S,HD]; V in [B,NH,HD,S].
// Grid: (S/128, NH, B); 256 threads = 8 waves; wave owns 16 query rows.
// K / Vt tiles staged via GLOBAL_LOAD_ASYNC_TO_LDS_B128 (ASYNCcnt).
// ---------------------------------------------------------------------------
__global__ __launch_bounds__(256)
void flash_attn_kernel(const __bf16* __restrict__ Q, const __bf16* __restrict__ K,
                       const __bf16* __restrict__ Vt, float* __restrict__ out) {
  __shared__ __bf16 sK[64][HD];        // [key][d]   8 KB
  __shared__ __bf16 sVt[HD][64];       // [d][key]   8 KB
  __shared__ __bf16 sP[8][16][64];     // per-wave P staging, 16 KB

  const int tid  = threadIdx.x;
  const int lane = tid & 31, wave = tid >> 5;
  const int lr   = lane & 15, lh = lane >> 4;
  const int h    = blockIdx.y, b = blockIdx.z;
  const int headoff = (b * NHD + h) * SS * HD;   // fits in int (max 12.58M)
  const __bf16* Qh  = Q + headoff;
  const __bf16* Kh  = K + headoff;
  const __bf16* Vth = Vt + headoff;    // [HD][SS] within head
  const int q0 = blockIdx.x * 128 + wave * 16;

  // Persistent Q fragments (16 rows x 64 dims -> 2 A-frags of 16x32)
  v16bf aQ[2];
#pragma unroll
  for (int s = 0; s < 2; ++s)
    aQ[s] = *(const v16bf*)(Qh + (q0 + lr) * HD + s * 32 + lh * 16);

  v8f acc[4] = {};
  float mrow[8], lsum[8];
#pragma unroll
  for (int r = 0; r < 8; ++r) { mrow[r] = -3.0e38f; lsum[r] = 0.f; }

  for (int kb = 0; kb < SS / 64; ++kb) {
    __syncthreads();   // previous iteration's LDS reads done
    // Async-stage K (row-major) and Vt (d-major): 512 x 16B segments each,
    // 2 per thread per tile.
#pragma unroll
    for (int i = 0; i < 2; ++i) {
      int s   = tid + i * 256;        // 0..511
      int row = s >> 3;               // 0..63 (key for K, d for Vt)
      int col = (s & 7) * 8;          // bf16 column, 16B granularity
      async_copy_b128((unsigned)(size_t)&sK[row][col],
                      Kh + (kb * 64 + row) * HD + col);
      async_copy_b128((unsigned)(size_t)&sVt[row][col],
                      Vth + row * SS + kb * 64 + col);
    }
    wait_async0();
    __syncthreads();

    // Scores: 16 q-rows x 64 keys = 4 C-tiles, contraction over HD=64
    v8f sc[4];
#pragma unroll
    for (int nt = 0; nt < 4; ++nt) {
      v8f s = {};
#pragma unroll
      for (int ks = 0; ks < 2; ++ks) {
        v16bf bk = *(const v16bf*)&sK[nt * 16 + lr][ks * 32 + lh * 16];
        s = wmma_bf16(aQ[ks], bk, s);
      }
      sc[nt] = s;
    }

    // Online softmax (scale 1/sqrt(64) = 0.125), per C-row (16-lane halves)
    float fac[8];
#pragma unroll
    for (int r = 0; r < 8; ++r) {
      float mx = -3.0e38f;
#pragma unroll
      for (int nt = 0; nt < 4; ++nt) {
        float x = sc[nt][r] * 0.125f;
        sc[nt][r] = x;
        mx = fmaxf(mx, x);
      }
#pragma unroll
      for (int off = 1; off < 16; off <<= 1)
        mx = fmaxf(mx, __shfl_xor(mx, off, 32));
      float mnew = fmaxf(mrow[r], mx);
      float f = __expf(mrow[r] - mnew);
      mrow[r] = mnew;
      fac[r] = f;
      float rsum = 0.f;
#pragma unroll
      for (int nt = 0; nt < 4; ++nt) {
        float p = __expf(sc[nt][r] - mnew);
        sc[nt][r] = p;
        rsum += p;
      }
#pragma unroll
      for (int off = 1; off < 16; off <<= 1)
        rsum += __shfl_xor(rsum, off, 32);
      lsum[r] = lsum[r] * f + rsum;
    }

    // Rescale ctx accumulators; stage P (C-layout -> A-layout via LDS)
#pragma unroll
    for (int dt = 0; dt < 4; ++dt)
#pragma unroll
      for (int r = 0; r < 8; ++r) acc[dt][r] *= fac[r];

#pragma unroll
    for (int nt = 0; nt < 4; ++nt)
#pragma unroll
      for (int r = 0; r < 8; ++r)
        sP[wave][r + 8 * lh][nt * 16 + lr] = (__bf16)sc[nt][r];

    asm volatile("s_wait_dscnt 0" ::: "memory");  // same-wave LDS RAW

    // ctx += P(16x64) x V(64x16-per-tile)
#pragma unroll
    for (int ks = 0; ks < 2; ++ks) {
      v16bf aP = *(const v16bf*)&sP[wave][lr][ks * 32 + lh * 16];
#pragma unroll
      for (int dt = 0; dt < 4; ++dt) {
        v16bf bv = *(const v16bf*)&sVt[dt * 16 + lr][ks * 32 + lh * 16];
        acc[dt] = wmma_bf16(aP, bv, acc[dt]);
      }
    }
  }

  // Normalize and store fp32 output [B, S, H]
#pragma unroll
  for (int dt = 0; dt < 4; ++dt) {
#pragma unroll
    for (int r = 0; r < 8; ++r) {
      int srow = q0 + r + 8 * lh;
      out[(size_t)(b * SS + srow) * HH + h * HD + dt * 16 + lr] =
          acc[dt][r] / lsum[r];
    }
  }
}

// ---------------------------------------------------------------------------
extern "C" void kernel_launch(void* const* d_in, const int* in_sizes, int n_in,
                              void* d_out, int out_size, void* d_ws, size_t ws_size,
                              hipStream_t stream) {
  const float* hs = (const float*)d_in[0];
  const float* Wq = (const float*)d_in[1];
  const float* Wk = (const float*)d_in[2];
  const float* Wv = (const float*)d_in[3];
  const float* bq = (const float*)d_in[4];
  const float* bk = (const float*)d_in[5];
  const float* bv = (const float*)d_in[6];
  float* out = (float*)d_out;

  const size_t w_elems    = (size_t)HH * HH;             // 589824
  const size_t head_elems = (size_t)BB * NHD * SS * HD;  // 12.58M

  __bf16* wqb = (__bf16*)d_ws;
  __bf16* wkb = wqb + w_elems;
  __bf16* wvb = wkb + w_elems;
  __bf16* qws = wvb + w_elems;
  __bf16* kws = qws + head_elems;
  __bf16* vws = kws + head_elems;   // stored transposed [B,NH,HD,S]

  // Pre-convert weights to bf16 (once per launch)
  {
    int n4 = (int)(w_elems / 4);                 // 147456
    dim3 g((n4 + 255) / 256);
    cvt_f32_to_bf16_kernel<<<g, 256, 0, stream>>>(Wq, wqb, n4);
    cvt_f32_to_bf16_kernel<<<g, 256, 0, stream>>>(Wk, wkb, n4);
    cvt_f32_to_bf16_kernel<<<g, 256, 0, stream>>>(Wv, wvb, n4);
  }

  dim3 gemm_grid(BB * SS / 32);   // 512
  qkv_proj_kernel<<<gemm_grid, 256, 0, stream>>>(hs, wqb, wkb, wvb,
                                                 bq, bk, bv, qws, kws, vws);

  dim3 attn_grid(SS / 128, NHD, BB);  // (8, 12, 16)
  flash_attn_kernel<<<attn_grid, 256, 0, stream>>>(qws, kws, vws, out);
}
